// CosineSimCodebook_60395830116470
// MI455X (gfx1250) — matrix operand order repaired and checked by reading, not compiled, not run on hardware
//
#include <hip/hip_runtime.h>
#include <hip/hip_bf16.h>
#include <stdint.h>
#include <stddef.h>

// ---- problem constants (from reference setup_inputs) ----
#define ROWS   8192        // B*N = 8*1024
#define DIM    256         // D
#define CODES  8192        // C
#define QELEMS (ROWS * DIM)

#define CHUNKS      16                     // code-dimension split for parallelism
#define CTILES      (CODES / 16)           // 512 column tiles total
#define CT_PER_CHK  (CTILES / CHUNKS)      // 32 column tiles per chunk

typedef __attribute__((ext_vector_type(16))) __bf16          v16bf;
typedef __attribute__((ext_vector_type(8)))  float           v8f;
typedef __attribute__((ext_vector_type(4)))  float           f4v;
typedef __attribute__((ext_vector_type(4)))  unsigned int    u4v;
typedef __attribute__((ext_vector_type(4)))  unsigned short  us4v;

__device__ __forceinline__ unsigned short f32_to_bf16_rne(float f) {
    unsigned int u = __float_as_uint(f);
    unsigned int r = u + 0x7FFFu + ((u >> 16) & 1u);   // round-to-nearest-even
    return (unsigned short)(r >> 16);
}

// ---------------------------------------------------------------------------
// Kernel 1: row-wise L2 normalize (x and embed) -> bf16 workspace
// One wave32 per 256-element row; 8 rows per 256-thread block.
// ---------------------------------------------------------------------------
__global__ void __launch_bounds__(256)
vq_normalize_bf16(const float* __restrict__ x,
                  const float* __restrict__ embed,
                  unsigned short* __restrict__ xn,
                  unsigned short* __restrict__ en)
{
    const int wave = (int)((blockIdx.x * blockDim.x + threadIdx.x) >> 5);
    const int lane = (int)(threadIdx.x & 31);

    const float* src;
    unsigned short* dst;
    if (wave < ROWS) {
        src = x + (size_t)wave * DIM;
        dst = xn + (size_t)wave * DIM;
    } else {
        src = embed + (size_t)(wave - ROWS) * DIM;
        dst = en + (size_t)(wave - ROWS) * DIM;
    }

    f4v a = *(const f4v*)(src + lane * 4);
    f4v b = *(const f4v*)(src + lane * 4 + 128);

    float s = a[0]*a[0] + a[1]*a[1] + a[2]*a[2] + a[3]*a[3]
            + b[0]*b[0] + b[1]*b[1] + b[2]*b[2] + b[3]*b[3];
    #pragma unroll
    for (int m = 16; m >= 1; m >>= 1)
        s += __shfl_xor(s, m, 32);

    const float inv = 1.0f / fmaxf(sqrtf(s), 1e-12f);

    us4v pa, pb;
    #pragma unroll
    for (int i = 0; i < 4; ++i) {
        pa[i] = f32_to_bf16_rne(a[i] * inv);
        pb[i] = f32_to_bf16_rne(b[i] * inv);
    }
    *(us4v*)(dst + lane * 4)       = pa;
    *(us4v*)(dst + lane * 4 + 128) = pb;
}

// ---------------------------------------------------------------------------
// Kernel 2: fused cosine-sim GEMM (bf16 WMMA, f32 acc) + running argmax.
//
// Grid: (32 row-groups, 16 code chunks); block = 8 waves.
// Wave w of block (gx, chunk) owns rows [ (gx*8+w)*32, +32 ) -- TWO 16-row
// tiles with resident A fragments -- and scans codes [chunk*512, +512).
// Every B fragment (1 KB) feeds TWO wmma ops. All 8 waves of a block stream
// the SAME B range -> L0 hits.
//
// Fragment layouts (v_wmma_f32_16x16x32_bf16, wave32), per ISA 7.12.2/7.12.4:
//   A (16x32, MxK): lane L -> row (L&15); elems 0..7  = K[kb + (L>>4)*8 .. +7]
//                                         elems 8..15 = K[kb + 16 + (L>>4)*8 .. +7]
//   B (32x16, KxN): lane L -> col (L&15); elems 0..7  = K[kb + (L>>4)*8 .. +7]
//                   elems 8..15 = K[kb + 16 + (L>>4)*8 .. +7] (same order as A)
//   D (16x16 f32):  elem v, lane L -> row (L>>4)*8+v, col (L&15)
// ---------------------------------------------------------------------------
__global__ void __launch_bounds__(256)
vq_dist_argmax(const unsigned short* __restrict__ xn,
               const unsigned short* __restrict__ en,
               float* __restrict__ partV,     // [ROWS][CHUNKS]
               int*   __restrict__ partI)     // [ROWS][CHUNKS]
{
    const int lane    = (int)(threadIdx.x & 31);
    const int wave    = (int)(threadIdx.x >> 5);
    const int rowPair = (int)blockIdx.x * 8 + wave;   // 0..255
    const int chunk   = (int)blockIdx.y;              // 0..15
    const int rowBase0 = rowPair * 32;
    const int rowBase1 = rowBase0 + 16;
    const int r    = lane & 15;
    const int half = lane >> 4;

    // Resident A fragments for both 16-row tiles (K = 256 -> 8 frags each)
    const unsigned short* arow0 = xn + (size_t)(rowBase0 + r) * DIM + half * 8;
    const unsigned short* arow1 = xn + (size_t)(rowBase1 + r) * DIM + half * 8;
    v16bf aF0[8], aF1[8];
    #pragma unroll
    for (int ks = 0; ks < 8; ++ks) {
        union { u4v u[2]; v16bf v; } t0, t1;
        t0.u[0] = *(const u4v*)(arow0 + ks * 32);
        t0.u[1] = *(const u4v*)(arow0 + ks * 32 + 16);
        t1.u[0] = *(const u4v*)(arow1 + ks * 32);
        t1.u[1] = *(const u4v*)(arow1 + ks * 32 + 16);
        aF0[ks] = t0.v;
        aF1[ks] = t1.v;
    }

    const int colBase = chunk * (CT_PER_CHK * 16);
    // B: lane holds column (L&15) = code row, K runs [half*8..+7] and [16+half*8..+7]
    const unsigned short* brow =
        en + (size_t)(colBase + r) * DIM + half * 8;

    float bestV0[8], bestV1[8];
    int   bestI0[8], bestI1[8];
    #pragma unroll
    for (int v = 0; v < 8; ++v) {
        bestV0[v] = -3.0e38f; bestI0[v] = 0;
        bestV1[v] = -3.0e38f; bestI1[v] = 0;
    }

    for (int ct = 0; ct < CT_PER_CHK; ++ct) {
        const unsigned short* bp = brow + (size_t)ct * 16 * DIM;
        v8f acc0 = {};
        v8f acc1 = {};
        #pragma unroll
        for (int ks = 0; ks < 8; ++ks) {
            union { u4v u[2]; v16bf v; } t;
            t.u[0] = *(const u4v*)(bp + ks * 32);       // K half*8 .. +7
            t.u[1] = *(const u4v*)(bp + ks * 32 + 16);  // K 16+half*8 .. +7
            acc0 = __builtin_amdgcn_wmma_f32_16x16x32_bf16(
                       false, aF0[ks], false, t.v, (short)0, acc0, false, false);
            acc1 = __builtin_amdgcn_wmma_f32_16x16x32_bf16(
                       false, aF1[ks], false, t.v, (short)0, acc1, false, false);
        }
        const int col = colBase + ct * 16 + r;
        #pragma unroll
        for (int v = 0; v < 8; ++v) {
            const float v0 = acc0[v];
            const float v1 = acc1[v];
            if (v0 > bestV0[v]) { bestV0[v] = v0; bestI0[v] = col; }
            if (v1 > bestV1[v]) { bestV1[v] = v1; bestI1[v] = col; }
        }
    }

    // Cross-lane reduce: lanes 0..15 hold rows base+v, lanes 16..31 base+8+v.
    // XOR masks 1,2,4,8 stay within each 16-lane half. Lower index wins ties.
    #pragma unroll
    for (int v = 0; v < 8; ++v) {
        float bv0 = bestV0[v]; int bi0 = bestI0[v];
        float bv1 = bestV1[v]; int bi1 = bestI1[v];
        #pragma unroll
        for (int m = 8; m >= 1; m >>= 1) {
            const float ov0 = __shfl_xor(bv0, m, 32);
            const int   oi0 = __shfl_xor(bi0, m, 32);
            const float ov1 = __shfl_xor(bv1, m, 32);
            const int   oi1 = __shfl_xor(bi1, m, 32);
            if (ov0 > bv0 || (ov0 == bv0 && oi0 < bi0)) { bv0 = ov0; bi0 = oi0; }
            if (ov1 > bv1 || (ov1 == bv1 && oi1 < bi1)) { bv1 = ov1; bi1 = oi1; }
        }
        if (r == 0) {
            const int row0 = rowBase0 + half * 8 + v;
            const int row1 = rowBase1 + half * 8 + v;
            partV[row0 * CHUNKS + chunk] = bv0;
            partI[row0 * CHUNKS + chunk] = bi0;
            partV[row1 * CHUNKS + chunk] = bv1;
            partI[row1 * CHUNKS + chunk] = bi1;
        }
    }
}

// ---------------------------------------------------------------------------
// Kernel 3: combine 16 per-chunk partials per row, then gather the
// UN-normalized codebook row into d_out and write idx (as float).
// One wave per row; 8 rows per 256-thread block.
// ---------------------------------------------------------------------------
__global__ void __launch_bounds__(256)
vq_combine_gather(const float* __restrict__ partV,
                  const int*   __restrict__ partI,
                  const float* __restrict__ embed,
                  float*       __restrict__ qout,
                  float*       __restrict__ idx_f)
{
    const int row  = (int)((blockIdx.x * blockDim.x + threadIdx.x) >> 5);
    const int lane = (int)(threadIdx.x & 31);

    float bv = -3.0e38f;
    int   bi = 0x7FFFFFFF;
    if (lane < CHUNKS) {
        bv = partV[row * CHUNKS + lane];
        bi = partI[row * CHUNKS + lane];
    }
    #pragma unroll
    for (int m = 8; m >= 1; m >>= 1) {      // reduce within lanes 0..15
        const float ov = __shfl_xor(bv, m, 32);
        const int   oi = __shfl_xor(bi, m, 32);
        if (ov > bv || (ov == bv && oi < bi)) { bv = ov; bi = oi; }
    }
    const int code = __shfl(bi, 0, 32);     // broadcast winner to all lanes

    const float* src = embed + (size_t)code * DIM;
    float*       dst = qout  + (size_t)row * DIM;
    *(f4v*)(dst + lane * 4)       = *(const f4v*)(src + lane * 4);
    *(f4v*)(dst + lane * 4 + 128) = *(const f4v*)(src + lane * 4 + 128);

    if (lane == 0)
        idx_f[row] = (float)code;
}

// ---------------------------------------------------------------------------
extern "C" void kernel_launch(void* const* d_in, const int* in_sizes, int n_in,
                              void* d_out, int out_size, void* d_ws, size_t ws_size,
                              hipStream_t stream)
{
    const float* x     = (const float*)d_in[0];   // [8,1024,256] f32
    const float* embed = (const float*)d_in[1];   // [1,8192,256] f32

    // workspace: xn bf16 (4MB) | en bf16 (4MB) | partV (512KB) | partI (512KB)
    unsigned short* xn    = (unsigned short*)d_ws;
    unsigned short* en    = xn + (size_t)ROWS * DIM;
    float*          partV = (float*)(en + (size_t)CODES * DIM);
    int*            partI = (int*)(partV + (size_t)ROWS * CHUNKS);

    float* qout  = (float*)d_out;          // quantize [8192,256] f32
    float* idx_f = qout + (size_t)QELEMS;  // idx as floats, concatenated

    vq_normalize_bf16<<<(ROWS + CODES) / 8, 256, 0, stream>>>(x, embed, xn, en);
    vq_dist_argmax<<<dim3(32, CHUNKS), 256, 0, stream>>>(xn, en, partV, partI);
    vq_combine_gather<<<ROWS / 8, 256, 0, stream>>>(partV, partI, embed, qout, idx_f);
}
